// CTMHead_52750788329565
// MI455X (gfx1250) — matrix operand order repaired and checked by reading, not compiled, not run on hardware
//
#include <hip/hip_runtime.h>
#include <hip/hip_bf16.h>
#include <math.h>

// ---------------------------------------------------------------------------
// CTM head for MI455X / gfx1250 (wave32, WMMA bf16 matrix cores).
// All GEMMs + attention run on v_wmma_f32_16x16x32_bf16; LN/softmax/NLM on VALU.
// ---------------------------------------------------------------------------

typedef __bf16 bf16_t;
typedef __attribute__((ext_vector_type(16))) __bf16 v16bf;
typedef __attribute__((ext_vector_type(8)))  float  v8f;

#define BB    4
#define SS    1024
#define L0C   2048
#define HIDC  512
#define DC    256
#define MC    25
#define DHC   32
#define PC    512
#define NHEAD 8
#define HDIM  64
#define TTK   10
#define ROWS  (BB*SS)          // 4096

__device__ __forceinline__ bf16_t f2bf(float f) {
  union { float f; unsigned u; } a; a.f = f;
  unsigned r = a.u + 0x7FFFu + ((a.u >> 16) & 1u);   // round-to-nearest-even
  unsigned short hs = (unsigned short)(r >> 16);
  return __builtin_bit_cast(bf16_t, hs);
}

// Build a v16bf fragment from two 16-byte chunks (8 bf16 each).
__device__ __forceinline__ v16bf load_frag(const bf16_t* p0, const bf16_t* p1) {
  union { uint4 u; bf16_t h[8]; } c0, c1;
  c0.u = *reinterpret_cast<const uint4*>(p0);
  c1.u = *reinterpret_cast<const uint4*>(p1);
  v16bf r;
#pragma unroll
  for (int i = 0; i < 8; ++i) { r[i] = c0.h[i]; r[8 + i] = c1.h[i]; }
  return r;
}

__device__ __forceinline__ v8f wmma_bf16(v16bf a, v16bf b, v8f c) {
  return __builtin_amdgcn_wmma_f32_16x16x32_bf16(false, a, false, b, (short)0, c,
                                                 false, false);
}

__device__ __forceinline__ float block_reduce_sum(float v, float* red) {
  int tid = threadIdx.x;
  red[tid] = v; __syncthreads();
  for (int s = 128; s > 0; s >>= 1) {
    if (tid < s) red[tid] += red[tid + s];
    __syncthreads();
  }
  float r = red[0]; __syncthreads();
  return r;
}

// ---------------------------------------------------------------------------
// Weight convert+transpose: W (K,N) f32 -> Wt (N,K) bf16
// ---------------------------------------------------------------------------
__global__ void wt_convert(const float* __restrict__ W, bf16_t* __restrict__ Wt,
                           int K, int N) {
  int i = blockIdx.x * 256 + threadIdx.x;
  if (i >= K * N) return;
  int k = i / N, n = i % N;
  Wt[(size_t)n * K + k] = f2bf(W[i]);
}

// ---------------------------------------------------------------------------
// Generic row LayerNorm (+ optional activation), f32 in -> f32/bf16 out
// act: 0 none, 1 tanh, 2 exact gelu
// ---------------------------------------------------------------------------
__global__ __launch_bounds__(256) void ln_rows(const float* __restrict__ X,
    const float* __restrict__ g, const float* __restrict__ b,
    float* __restrict__ of, bf16_t* __restrict__ oh, int C, int act) {
  __shared__ float red[256];
  const int tid = threadIdx.x;
  const float* x = X + (size_t)blockIdx.x * C;
  float s = 0.f;
  for (int c = tid; c < C; c += 256) s += x[c];
  float mu = block_reduce_sum(s, red) / (float)C;
  float v = 0.f;
  for (int c = tid; c < C; c += 256) { float d = x[c] - mu; v += d * d; }
  float var = block_reduce_sum(v, red) / (float)C;
  float inv = rsqrtf(var + 1e-5f);
  for (int c = tid; c < C; c += 256) {
    float y = (x[c] - mu) * inv * g[c] + b[c];
    if (act == 1) y = tanhf(y);
    else if (act == 2) y = 0.5f * y * (1.f + erff(y * 0.70710678118654752f));
    size_t idx = (size_t)blockIdx.x * C + c;
    if (of) of[idx] = y;
    if (oh) oh[idx] = f2bf(y);
  }
}

// ---------------------------------------------------------------------------
// bf16 WMMA GEMM: C[4096 x N] = act(A[4096 x K] @ Wt[N x K]^T + bias)
// 128x64 block tile, 8 waves of 32x32, BK=32.
// ---------------------------------------------------------------------------
#define GTM 128
#define GTN 64
#define GTK 32
#define LPAD 8

__global__ __launch_bounds__(256) void gemm_bf16_kernel(
    const bf16_t* __restrict__ A, const bf16_t* __restrict__ Wt,
    const float* __restrict__ bias, float* __restrict__ Cf,
    bf16_t* __restrict__ Ch, int N, int K, int act) {
  __shared__ alignas(16) bf16_t As[GTM][GTK + LPAD];
  __shared__ alignas(16) bf16_t Bs[GTN][GTK + LPAD];
  const int tid = threadIdx.x;
  const int lane = tid & 31, wave = tid >> 5;
  const int wy = wave >> 1, wx = wave & 1;            // 4 x 2 wave grid
  const int rowBase = blockIdx.y * GTM;
  const int colBase = blockIdx.x * GTN;
  const int half = lane >> 4, l16 = lane & 15;

  v8f acc[2][2];
#pragma unroll
  for (int i = 0; i < 2; ++i)
#pragma unroll
    for (int j = 0; j < 2; ++j) acc[i][j] = v8f{};

  for (int k0 = 0; k0 < K; k0 += GTK) {
    for (int i = tid; i < (GTM * GTK) / 8; i += 256) {
      int r = i >> 2, c = (i & 3) * 8;
      *reinterpret_cast<uint4*>(&As[r][c]) =
          *reinterpret_cast<const uint4*>(&A[(size_t)(rowBase + r) * K + k0 + c]);
    }
    for (int i = tid; i < (GTN * GTK) / 8; i += 256) {
      int r = i >> 2, c = (i & 3) * 8;
      *reinterpret_cast<uint4*>(&Bs[r][c]) =
          *reinterpret_cast<const uint4*>(&Wt[(size_t)(colBase + r) * K + k0 + c]);
    }
    __syncthreads();
    // A fragments (rows), 16-bit A layout: K = half*16 .. contiguous
    v16bf a0 = load_frag(&As[wy * 32 + l16][half * 16],
                         &As[wy * 32 + l16][half * 16 + 8]);
    v16bf a1 = load_frag(&As[wy * 32 + 16 + l16][half * 16],
                         &As[wy * 32 + 16 + l16][half * 16 + 8]);
    // B fragments (cols), same k order as A: K = half*16 .. contiguous
    v16bf b0 = load_frag(&Bs[wx * 32 + l16][half * 16],
                         &Bs[wx * 32 + l16][half * 16 + 8]);
    v16bf b1 = load_frag(&Bs[wx * 32 + 16 + l16][half * 16],
                         &Bs[wx * 32 + 16 + l16][half * 16 + 8]);
    acc[0][0] = wmma_bf16(a0, b0, acc[0][0]);
    acc[0][1] = wmma_bf16(a0, b1, acc[0][1]);
    acc[1][0] = wmma_bf16(a1, b0, acc[1][0]);
    acc[1][1] = wmma_bf16(a1, b1, acc[1][1]);
    __syncthreads();
  }

#pragma unroll
  for (int i = 0; i < 2; ++i)
#pragma unroll
    for (int j = 0; j < 2; ++j)
#pragma unroll
      for (int r = 0; r < 8; ++r) {
        int row = rowBase + wy * 32 + i * 16 + r + 8 * half;
        int col = colBase + wx * 32 + j * 16 + l16;
        float vv = acc[i][j][r] + bias[col];
        if (act == 1) vv = tanhf(vv);
        size_t idx = (size_t)row * N + col;
        if (Cf) Cf[idx] = vv;
        if (Ch) Ch[idx] = f2bf(vv);
      }
}

// ---------------------------------------------------------------------------
// Flash attention: one wave per 16-row q tile; causal; bf16 WMMA both matmuls.
// Q,K: (B*S, HID) bf16 row-major; Vt: (B,H,64,S) bf16; out AO: (B*S,HID)
// ---------------------------------------------------------------------------
__global__ __launch_bounds__(256) void attn_kernel(
    const bf16_t* __restrict__ Q, const bf16_t* __restrict__ Kb,
    const bf16_t* __restrict__ Vt, float* __restrict__ AOf,
    bf16_t* __restrict__ AOh) {
  __shared__ alignas(16) bf16_t pshare[8][16 * 32];
  const int wave = threadIdx.x >> 5, lane = threadIdx.x & 31;
  const int w = blockIdx.x * 8 + wave;         // 2048 q tiles total
  const int qt = w & 63, h = (w >> 6) & 7, b = w >> 9;
  const int qbase = qt * 16;
  const int l16 = lane & 15, half = lane >> 4;
  const int bs0 = b * SS;
  bf16_t* pl = &pshare[wave][0];

  const bf16_t* qr = Q + ((size_t)(bs0 + qbase + l16) * HIDC + h * HDIM);
  v16bf qa0 = load_frag(qr + half * 16, qr + half * 16 + 8);         // hd 0..31
  v16bf qa1 = load_frag(qr + 32 + half * 16, qr + 40 + half * 16);    // hd 32..63

  v8f acc[4];
#pragma unroll
  for (int nt = 0; nt < 4; ++nt) acc[nt] = v8f{};
  float mrow[8], lrow[8];
#pragma unroll
  for (int r = 0; r < 8; ++r) { mrow[r] = -3e38f; lrow[r] = 0.f; }

  const int nblk = (qbase >> 5) + 1;
  for (int j = 0; j < nblk; ++j) {
    const int kb = j * 32;
    // ---- scores: Q(16x64) . K^T -> 16x32 (two 16x16 tiles)
    v8f s0 = v8f{}, s1 = v8f{};
    {
      const bf16_t* kr = Kb + ((size_t)(bs0 + kb + l16) * HIDC + h * HDIM);
      v16bf kf0 = load_frag(kr + half * 16, kr + half * 16 + 8);
      v16bf kf1 = load_frag(kr + 32 + half * 16, kr + 32 + half * 16 + 8);
      s0 = wmma_bf16(qa0, kf0, s0);
      s0 = wmma_bf16(qa1, kf1, s0);
    }
    {
      const bf16_t* kr = Kb + ((size_t)(bs0 + kb + 16 + l16) * HIDC + h * HDIM);
      v16bf kf0 = load_frag(kr + half * 16, kr + half * 16 + 8);
      v16bf kf1 = load_frag(kr + 32 + half * 16, kr + 32 + half * 16 + 8);
      s1 = wmma_bf16(qa0, kf0, s1);
      s1 = wmma_bf16(qa1, kf1, s1);
    }
    // ---- scale + causal mask + online softmax
    float fac[8];
#pragma unroll
    for (int r = 0; r < 8; ++r) {
      int row = qbase + r + 8 * half;
      float v0 = s0[r] * 0.125f; if (kb + l16 > row) v0 = -1e9f;
      float v1 = s1[r] * 0.125f; if (kb + 16 + l16 > row) v1 = -1e9f;
      float mx = fmaxf(v0, v1);
#pragma unroll
      for (int o = 8; o >= 1; o >>= 1) mx = fmaxf(mx, __shfl_xor(mx, o, 32));
      mx = fmaxf(mx, mrow[r]);
      fac[r] = __expf(mrow[r] - mx);
      mrow[r] = mx;
      float p0 = __expf(v0 - mx), p1 = __expf(v1 - mx);
      pl[(r + 8 * half) * 32 + l16] = f2bf(p0);
      pl[(r + 8 * half) * 32 + 16 + l16] = f2bf(p1);
      float ps = p0 + p1;
#pragma unroll
      for (int o = 8; o >= 1; o >>= 1) ps += __shfl_xor(ps, o, 32);
      lrow[r] = lrow[r] * fac[r] + ps;
    }
#pragma unroll
    for (int nt = 0; nt < 4; ++nt)
#pragma unroll
      for (int r = 0; r < 8; ++r) acc[nt][r] *= fac[r];
    // wave-local LDS RAW: wait DS ops (split counters on gfx1250)
    asm volatile("s_wait_dscnt 0" ::: "memory");
    v16bf pa = load_frag(pl + l16 * 32 + half * 16,
                         pl + l16 * 32 + half * 16 + 8);
    // ---- P(16x32) . V(32x64): Vt is hd-major so B-frags are contiguous
    const bf16_t* vtb = Vt + (size_t)((b * NHEAD + h) * HDIM) * SS + kb;
#pragma unroll
    for (int nt = 0; nt < 4; ++nt) {
      const bf16_t* vr = vtb + (size_t)(nt * 16 + l16) * SS;
      v16bf vf = load_frag(vr + half * 16, vr + half * 16 + 8);
      acc[nt] = wmma_bf16(pa, vf, acc[nt]);
    }
  }
  // ---- finalize
#pragma unroll
  for (int nt = 0; nt < 4; ++nt)
#pragma unroll
    for (int r = 0; r < 8; ++r) {
      int row = qbase + r + 8 * half;
      int col = h * HDIM + nt * 16 + l16;
      size_t idx = (size_t)(bs0 + row) * HIDC + col;
      float ov = acc[nt][r] / lrow[r];
      AOh[idx] = f2bf(ov);
      if (AOf) AOf[idx] = ov;
    }
}

// ---------------------------------------------------------------------------
// Elementwise kernels
// ---------------------------------------------------------------------------
__global__ void zst_init(const float* __restrict__ z0, float* __restrict__ zst) {
  int i = blockIdx.x * 256 + threadIdx.x;
  if (i < ROWS * DC) zst[i] = z0[i % DC];
}

__global__ void sync_update(const float* __restrict__ zst,
                            const int* __restrict__ il, const int* __restrict__ ir,
                            const float* __restrict__ rraw,
                            float* __restrict__ alpha, float* __restrict__ beta,
                            bf16_t* __restrict__ sync_h, int t) {
  int idx = blockIdx.x * 256 + threadIdx.x;
  if (idx >= ROWS * PC) return;
  int p = idx & (PC - 1), bs = idx >> 9;
  float g = 1.f / (1.f + __expf(-rraw[p]));
  float prod = zst[(size_t)bs * DC + il[p]] * zst[(size_t)bs * DC + ir[p]];
  float a_old = t ? alpha[idx] : 0.f;
  float b_old = t ? beta[idx] : 1.f;
  float a = g * a_old + (1.f - g) * prod;
  float bb = g * b_old + (1.f - g);
  alpha[idx] = a; beta[idx] = bb;
  sync_h[idx] = f2bf(a / fmaxf(bb, 1e-4f));
}

__global__ void concat_syn(const float* __restrict__ zst,
                           const float* __restrict__ o,
                           bf16_t* __restrict__ syn_in) {
  int i = blockIdx.x * 256 + threadIdx.x;
  if (i >= ROWS * (DC + HIDC)) return;
  int row = i / (DC + HIDC), c = i % (DC + HIDC);
  float v = (c < DC) ? zst[(size_t)row * DC + c]
                     : o[(size_t)row * HIDC + (c - DC)];
  syn_in[i] = f2bf(v);
}

// v bf16 (B*S, HID) -> Vt bf16 (B, H, hd, S)
__global__ void v_transpose(const bf16_t* __restrict__ v, bf16_t* __restrict__ vt) {
  int i = blockIdx.x * 256 + threadIdx.x;
  if (i >= ROWS * HIDC) return;
  int sg = i / HIDC, c = i % HIDC;
  int b = sg / SS, s = sg % SS, h = c / HDIM, hd = c % HDIM;
  vt[(size_t)((b * NHEAD + h) * HDIM + hd) * SS + s] = v[i];
}

// ---------------------------------------------------------------------------
// Per-neuron NLM: one block per (b,s); thread d does DH=32 matvecs over M=25,
// then 8192-wide LN + tanh + DH contraction.
// ---------------------------------------------------------------------------
__global__ __launch_bounds__(256) void nlm_kernel(
    const float* __restrict__ a_init, const float* __restrict__ a_store,
    const float* __restrict__ w1, const float* __restrict__ b1,
    const float* __restrict__ lng, const float* __restrict__ lnb,
    const float* __restrict__ w2, const float* __restrict__ b2,
    float* __restrict__ zst, int t) {
  __shared__ float red[256];
  const int bs = blockIdx.x, d = threadIdx.x;
  float a[MC];
#pragma unroll
  for (int j = 0; j < MC; ++j) {
    int lag = (MC - 1) - j;                       // newest at j = M-1
    if (lag <= t)
      a[j] = a_store[(size_t)(t - lag) * (ROWS * DC) + (size_t)bs * DC + d];
    else
      a[j] = a_init[d * MC + (j + t + 1)];
  }
  const float sc = 0.2f;                          // 1/sqrt(M)
  float nh[DHC];
  float sum = 0.f, sq = 0.f;
#pragma unroll 4
  for (int h = 0; h < DHC; ++h) {
    float accv = b1[h * DC + d];
#pragma unroll
    for (int m = 0; m < MC; ++m)
      accv += (a[m] * sc) * w1[(m * DHC + h) * DC + d];
    nh[h] = accv; sum += accv; sq += accv * accv;
  }
  float S1 = block_reduce_sum(sum, red);
  float S2 = block_reduce_sum(sq, red);
  const float n = (float)(DHC * DC);
  float mu = S1 / n;
  float var = S2 / n - mu * mu;
  float inv = rsqrtf(var + 1e-5f);
  float z = b2[d];
#pragma unroll 4
  for (int h = 0; h < DHC; ++h) {
    float hv = tanhf((nh[h] - mu) * inv * lng[h * DC + d] + lnb[h * DC + d]);
    z += hv * w2[h * DC + d];
  }
  zst[(size_t)bs * DC + d] = tanhf(z);
}

// ---------------------------------------------------------------------------
// certainty / value / synchronization heads: one wave per row
// ---------------------------------------------------------------------------
__global__ __launch_bounds__(256) void head_small(
    const float* __restrict__ hout, const float* __restrict__ zst,
    const float* __restrict__ certW, const float* __restrict__ certb,
    const float* __restrict__ valW, const float* __restrict__ valb,
    float* __restrict__ out_sync, float* __restrict__ out_cert,
    float* __restrict__ out_val) {
  int wave = threadIdx.x >> 5, lane = threadIdx.x & 31;
  int row = blockIdx.x * 8 + wave;
  const float* hr = hout + (size_t)row * HIDC;
  float c = 0.f, v = 0.f;
  for (int i = lane; i < HIDC; i += 32) {
    float h = hr[i]; c += h * certW[i]; v += h * valW[i];
  }
  float s = 0.f;
  const float* zr = zst + (size_t)row * DC;
  for (int i = lane; i < DC; i += 32) s += fabsf(zr[i]);
#pragma unroll
  for (int o = 16; o >= 1; o >>= 1) {
    c += __shfl_xor(c, o, 32);
    v += __shfl_xor(v, o, 32);
    s += __shfl_xor(s, o, 32);
  }
  if (lane == 0) {
    out_cert[row] = 1.f / (1.f + __expf(-(c + certb[0])));
    out_val[row]  = v + valb[0];
    out_sync[row] = s / (float)DC;
  }
}

// ---------------------------------------------------------------------------
// Host orchestration
// ---------------------------------------------------------------------------
extern "C" void kernel_launch(void* const* d_in, const int* in_sizes, int n_in,
                              void* d_out, int out_size, void* d_ws, size_t ws_size,
                              hipStream_t stream) {
  (void)in_sizes; (void)n_in; (void)out_size; (void)ws_size;
  const float* l0_seq  = (const float*)d_in[0];
  const float* in_g    = (const float*)d_in[1];
  const float* in_b    = (const float*)d_in[2];
  const float* kvW     = (const float*)d_in[3];
  const float* kvb     = (const float*)d_in[4];
  const float* kW      = (const float*)d_in[5];
  const float* kb_     = (const float*)d_in[6];
  const float* vW      = (const float*)d_in[7];
  const float* vb      = (const float*)d_in[8];
  const float* qW      = (const float*)d_in[9];
  const float* qb      = (const float*)d_in[10];
  const float* oaW     = (const float*)d_in[11];
  const float* oab     = (const float*)d_in[12];
  const float* syn_w1  = (const float*)d_in[13];
  const float* syn_b1  = (const float*)d_in[14];
  const float* syn_lg  = (const float*)d_in[15];
  const float* syn_lb  = (const float*)d_in[16];
  const float* syn_w2  = (const float*)d_in[17];
  const float* syn_b2  = (const float*)d_in[18];
  const float* nlm_w1  = (const float*)d_in[19];
  const float* nlm_b1  = (const float*)d_in[20];
  const float* nlm_lg  = (const float*)d_in[21];
  const float* nlm_lb  = (const float*)d_in[22];
  const float* nlm_w2  = (const float*)d_in[23];
  const float* nlm_b2  = (const float*)d_in[24];
  const float* ra_raw  = (const float*)d_in[25];
  const float* ro_raw  = (const float*)d_in[26];
  const float* op1W    = (const float*)d_in[27];
  const float* op1b    = (const float*)d_in[28];
  const float* op_lg   = (const float*)d_in[29];
  const float* op_lb   = (const float*)d_in[30];
  const float* op2W    = (const float*)d_in[31];
  const float* op2b    = (const float*)d_in[32];
  const float* fin_lg  = (const float*)d_in[33];
  const float* fin_lb  = (const float*)d_in[34];
  const float* certW   = (const float*)d_in[35];
  const float* certb   = (const float*)d_in[36];
  const float* valW    = (const float*)d_in[37];
  const float* valb    = (const float*)d_in[38];
  const float* z_init  = (const float*)d_in[39];
  const float* a_init  = (const float*)d_in[40];
  const int*   ia_l    = (const int*)d_in[41];
  const int*   ia_r    = (const int*)d_in[42];
  const int*   io_l    = (const int*)d_in[43];
  const int*   io_r    = (const int*)d_in[44];

  float* out = (float*)d_out;
  float* out_feat = out;                                 // (B,S,L0)
  float* out_sync = out + (size_t)ROWS * L0C;            // (B,S)
  float* out_cert = out_sync + ROWS;                     // (B,S,1)
  float* out_val  = out_cert + ROWS;                     // (B,S,1)

  // ---- bump-allocate workspace (assumes ~300 MB of scratch available)
  char* wsp = (char*)d_ws;
  auto alloc = [&](size_t bytes) -> void* {
    void* p = (void*)wsp;
    wsp += (bytes + 255) & ~(size_t)255;
    return p;
  };
  auto ab = [&](size_t elems) { return (bf16_t*)alloc(elems * sizeof(bf16_t)); };
  auto af = [&](size_t elems) { return (float*)alloc(elems * sizeof(float)); };

  bf16_t* xln    = ab((size_t)ROWS * L0C);
  bf16_t* kvWt   = ab((size_t)HIDC * L0C);
  bf16_t* kWt    = ab((size_t)HIDC * HIDC);
  bf16_t* vWt    = ab((size_t)HIDC * HIDC);
  bf16_t* qWt    = ab((size_t)HIDC * PC);
  bf16_t* oaWt   = ab((size_t)HIDC * HIDC);
  bf16_t* s1Wt   = ab((size_t)HIDC * (DC + HIDC));
  bf16_t* s2Wt   = ab((size_t)DC * HIDC);
  bf16_t* o1Wt   = ab((size_t)HIDC * PC);
  bf16_t* o2Wt   = ab((size_t)L0C * HIDC);
  bf16_t* kvh    = ab((size_t)ROWS * HIDC);
  bf16_t* kh     = ab((size_t)ROWS * HIDC);
  bf16_t* vh     = ab((size_t)ROWS * HIDC);
  bf16_t* vth    = ab((size_t)ROWS * HIDC);
  bf16_t* syncah = ab((size_t)ROWS * PC);
  bf16_t* syncoh = ab((size_t)ROWS * PC);
  bf16_t* qh     = ab((size_t)ROWS * HIDC);
  bf16_t* aoh    = ab((size_t)ROWS * HIDC);
  bf16_t* h1h    = ab((size_t)ROWS * HIDC);
  bf16_t* houth  = ab((size_t)ROWS * HIDC);
  bf16_t* synin  = ab((size_t)ROWS * (DC + HIDC));
  float*  zst    = af((size_t)ROWS * DC);
  float*  alpha_a = af((size_t)ROWS * PC);
  float*  beta_a  = af((size_t)ROWS * PC);
  float*  alpha_o = af((size_t)ROWS * PC);
  float*  beta_o  = af((size_t)ROWS * PC);
  float*  of      = af((size_t)ROWS * HIDC);
  float*  h1pre   = af((size_t)ROWS * HIDC);
  float*  astore  = af((size_t)TTK * ROWS * DC);
  float*  hpre    = af((size_t)ROWS * HIDC);
  float*  houtf   = af((size_t)ROWS * HIDC);
  float*  featpre = af((size_t)ROWS * L0C);

  auto ewg = [](size_t n) { return dim3((unsigned)((n + 255) / 256)); };
  auto gemm = [&](const bf16_t* A, const bf16_t* Wt, const float* bias,
                  float* Cf, bf16_t* Ch, int N, int K, int act) {
    dim3 grid(N / GTN, ROWS / GTM);
    gemm_bf16_kernel<<<grid, 256, 0, stream>>>(A, Wt, bias, Cf, Ch, N, K, act);
  };

  // ---- weights -> bf16 (N,K)
  wt_convert<<<ewg((size_t)L0C * HIDC), 256, 0, stream>>>(kvW, kvWt, L0C, HIDC);
  wt_convert<<<ewg((size_t)HIDC * HIDC), 256, 0, stream>>>(kW, kWt, HIDC, HIDC);
  wt_convert<<<ewg((size_t)HIDC * HIDC), 256, 0, stream>>>(vW, vWt, HIDC, HIDC);
  wt_convert<<<ewg((size_t)PC * HIDC), 256, 0, stream>>>(qW, qWt, PC, HIDC);
  wt_convert<<<ewg((size_t)HIDC * HIDC), 256, 0, stream>>>(oaW, oaWt, HIDC, HIDC);
  wt_convert<<<ewg((size_t)(DC + HIDC) * HIDC), 256, 0, stream>>>(syn_w1, s1Wt, DC + HIDC, HIDC);
  wt_convert<<<ewg((size_t)HIDC * DC), 256, 0, stream>>>(syn_w2, s2Wt, HIDC, DC);
  wt_convert<<<ewg((size_t)PC * HIDC), 256, 0, stream>>>(op1W, o1Wt, PC, HIDC);
  wt_convert<<<ewg((size_t)HIDC * L0C), 256, 0, stream>>>(op2W, o2Wt, HIDC, L0C);

  // ---- input LN, shared k/v
  ln_rows<<<ROWS, 256, 0, stream>>>(l0_seq, in_g, in_b, nullptr, xln, L0C, 0);
  gemm(xln, kvWt, kvb, nullptr, kvh, HIDC, L0C, 0);
  gemm(kvh, kWt, kb_, nullptr, kh, HIDC, HIDC, 0);
  gemm(kvh, vWt, vb, nullptr, vh, HIDC, HIDC, 0);
  v_transpose<<<ewg((size_t)ROWS * HIDC), 256, 0, stream>>>(vh, vth);
  zst_init<<<ewg((size_t)ROWS * DC), 256, 0, stream>>>(z_init, zst);

  // ---- recurrent ticks
  for (int t = 0; t < TTK; ++t) {
    sync_update<<<ewg((size_t)ROWS * PC), 256, 0, stream>>>(
        zst, ia_l, ia_r, ra_raw, alpha_a, beta_a, syncah, t);
    gemm(syncah, qWt, qb, nullptr, qh, HIDC, PC, 0);
    attn_kernel<<<256, 256, 0, stream>>>(qh, kh, vth, nullptr, aoh);
    gemm(aoh, oaWt, oab, of, nullptr, HIDC, HIDC, 0);
    concat_syn<<<ewg((size_t)ROWS * (DC + HIDC)), 256, 0, stream>>>(zst, of, synin);
    gemm(synin, s1Wt, syn_b1, h1pre, nullptr, HIDC, DC + HIDC, 0);
    ln_rows<<<ROWS, 256, 0, stream>>>(h1pre, syn_lg, syn_lb, nullptr, h1h, HIDC, 1);
    gemm(h1h, s2Wt, syn_b2, astore + (size_t)t * ROWS * DC, nullptr, DC, HIDC, 1);
    nlm_kernel<<<ROWS, DC, 0, stream>>>(a_init, astore, nlm_w1, nlm_b1, nlm_lg,
                                        nlm_lb, nlm_w2, nlm_b2, zst, t);
    sync_update<<<ewg((size_t)ROWS * PC), 256, 0, stream>>>(
        zst, io_l, io_r, ro_raw, alpha_o, beta_o, syncoh, t);
  }

  // ---- output heads
  gemm(syncoh, o1Wt, op1b, hpre, nullptr, HIDC, PC, 0);
  ln_rows<<<ROWS, 256, 0, stream>>>(hpre, op_lg, op_lb, houtf, houth, HIDC, 2);
  gemm(houth, o2Wt, op2b, featpre, nullptr, L0C, HIDC, 0);
  ln_rows<<<ROWS, 256, 0, stream>>>(featpre, fin_lg, fin_lb, out_feat, nullptr, L0C, 0);
  head_small<<<ROWS / 8, 256, 0, stream>>>(houtf, zst, certW, certb, valW, valb,
                                           out_sync, out_cert, out_val);
}
